// Proj2ScenePoint_33088428049085
// MI455X (gfx1250) — matrix operand order repaired and checked by reading, not compiled, not run on hardware
//
#include <hip/hip_runtime.h>
#include <hip/hip_bf16.h>

// ---------------- problem constants ----------------
constexpr int cN_PROJ = 400000;
constexpr int cN_SP   = 40000;
constexpr int cD      = 256;
constexpr int cH      = 8;
constexpr float cEPS  = 1e-5f;
constexpr float cNEG  = 0.2f;

// ---------------- vector types ----------------
typedef __attribute__((ext_vector_type(16))) __bf16 v16bf;
typedef __attribute__((ext_vector_type(8)))  __bf16 v8bf;
typedef __attribute__((ext_vector_type(8)))  float  v8f;
typedef __attribute__((ext_vector_type(4)))  float  v4f;

// LDS layout constants (padded strides to avoid bank conflicts)
constexpr int BSTR = 264;               // Bs row stride in bf16 elems (256 + 8 => 16B pad)
constexpr int STG  = 260;               // stage row stride in f32 elems (256 + 4 => 16B pad)
constexpr int SMEM_BYTES = 256 * BSTR * 2 + 8 * 16 * STG * 4;

// =====================================================================
// Weight transpose + f32 -> bf16:  WT[n][k] = W[k][n]
// =====================================================================
__global__ void transpose_bf16_kernel(const float* __restrict__ W0,
                                      const float* __restrict__ W1,
                                      const float* __restrict__ W2,
                                      __bf16* __restrict__ T0,
                                      __bf16* __restrict__ T1,
                                      __bf16* __restrict__ T2) {
    int bid = blockIdx.x;          // 0..767
    int mat = bid >> 8;
    int n   = bid & 255;
    int k   = threadIdx.x;
    const float* W = (mat == 0) ? W0 : (mat == 1) ? W1 : W2;
    __bf16*      T = (mat == 0) ? T0 : (mat == 1) ? T1 : T2;
    T[n * 256 + k] = (__bf16)W[k * 256 + n];
}

// =====================================================================
// LayerNorm + ReLU -> bf16, one wave per row (wave32)
// =====================================================================
__global__ void ln_relu_bf16_kernel(const float* __restrict__ X,
                                    const float* __restrict__ g,
                                    const float* __restrict__ b,
                                    __bf16* __restrict__ Y, int M) {
    int gw   = (int)((blockIdx.x * blockDim.x + threadIdx.x) >> 5);
    int lane = threadIdx.x & 31;
    if (gw >= M) return;
    const float* x = X + (size_t)gw * cD;
    int d0 = lane * 8;
    float v[8];
    float s = 0.f;
#pragma unroll
    for (int j = 0; j < 8; ++j) { v[j] = x[d0 + j]; s += v[j]; }
#pragma unroll
    for (int m = 16; m >= 1; m >>= 1) s += __shfl_xor(s, m, 32);
    float mean = s * (1.0f / 256.0f);
    float q = 0.f;
#pragma unroll
    for (int j = 0; j < 8; ++j) { float d = v[j] - mean; q += d * d; }
#pragma unroll
    for (int m = 16; m >= 1; m >>= 1) q += __shfl_xor(q, m, 32);
    float rstd = rsqrtf(q * (1.0f / 256.0f) + cEPS);
    v8bf o;
#pragma unroll
    for (int j = 0; j < 8; ++j) {
        float y = (v[j] - mean) * rstd * g[d0 + j] + b[d0 + j];
        y = y > 0.f ? y : 0.f;
        o[j] = (__bf16)y;
    }
    *(v8bf*)(Y + (size_t)gw * cD + d0) = o;
}

// =====================================================================
// WMMA GEMM: Out[M,256] = A[M,256] @ W[256,256] + bias
//   WT is W transposed ([n][k]) in bf16 (matches B-operand per-lane layout).
//   A_F32: A is f32 (converted to bf16 on the fly) else bf16.
//   EP: 0 = f32 out, 1 = bf16 out, 2 = f32 out += x_skip (residual)
// Block = 256 threads = 8 waves; each wave owns 16 rows x 256 cols.
// =====================================================================
template <bool A_F32, int EP>
__global__ __launch_bounds__(256) void gemm256_kernel(
        const void* __restrict__ Aptr,
        const __bf16* __restrict__ WT,
        const float* __restrict__ bias,
        void* __restrict__ Out,
        const float* __restrict__ xskip,
        int M) {
    extern __shared__ char smem[];
    __bf16* Bs       = (__bf16*)smem;                       // [256][BSTR]
    float*  stageAll = (float*)(smem + 256 * BSTR * 2);     // 8 x [16][STG]

    const int tid  = threadIdx.x;
    const int lane = tid & 31;
    const int wave = tid >> 5;

    // ---- stage WT (64K bf16) into padded LDS ----
#pragma unroll 4
    for (int i = 0; i < 32; ++i) {
        *(v8bf*)(Bs + tid * BSTR + i * 8) = *(const v8bf*)(WT + tid * 256 + i * 8);
    }
    __syncthreads();

    const int row0 = blockIdx.x * 128 + wave * 16;
    if (row0 >= M) return;                 // wave-uniform (M % 16 == 0 for all uses)

    float* stage = stageAll + wave * (16 * STG);

    const int mA   = lane & 15;            // A row within 16-row tile
    const int ksel = (lane >> 4) * 8;      // K sub-chunk select (A layout)

    // ---- load A block: 16 rows x 256 K, packed per ISA 16-bit A layout ----
    v16bf a[8];
    if (A_F32) {
        const float* Arow = (const float*)Aptr + (size_t)(row0 + mA) * 256;
        __builtin_prefetch(Arow + (size_t)128 * 256, 0, 1);   // next block's rows
#pragma unroll
        for (int kc = 0; kc < 8; ++kc) {
            const float* p = Arow + kc * 32 + ksel;
            v4f f0 = *(const v4f*)(p);
            v4f f1 = *(const v4f*)(p + 4);
            v4f f2 = *(const v4f*)(p + 16);
            v4f f3 = *(const v4f*)(p + 20);
            v16bf t;
#pragma unroll
            for (int j = 0; j < 4; ++j) {
                t[j]      = (__bf16)f0[j];
                t[4 + j]  = (__bf16)f1[j];
                t[8 + j]  = (__bf16)f2[j];
                t[12 + j] = (__bf16)f3[j];
            }
            a[kc] = t;
        }
    } else {
        const __bf16* Arow = (const __bf16*)Aptr + (size_t)(row0 + mA) * 256;
        __builtin_prefetch(Arow + (size_t)128 * 256, 0, 1);
#pragma unroll
        for (int kc = 0; kc < 8; ++kc) {
            const __bf16* p = Arow + kc * 32 + ksel;
            v8bf b0 = *(const v8bf*)(p);
            v8bf b1 = *(const v8bf*)(p + 16);
            v16bf t;
#pragma unroll
            for (int j = 0; j < 8; ++j) { t[j] = b0[j]; t[8 + j] = b1[j]; }
            a[kc] = t;
        }
    }

    const int nlane = lane & 15;           // B column within 16-col tile
    const int kh16  = (lane >> 4) * 8;     // B K sub-chunk select
    const int mbase = (lane >> 4) * 8;     // C/D row base

    // ---- 16 n-tiles x 8 k-steps of v_wmma_f32_16x16x32_bf16 ----
#pragma unroll 1
    for (int nt = 0; nt < 16; ++nt) {
        float bv = bias[nt * 16 + nlane];
        v8f acc = {bv, bv, bv, bv, bv, bv, bv, bv};
        const __bf16* brow = Bs + (nt * 16 + nlane) * BSTR + kh16;
#pragma unroll
        for (int kc = 0; kc < 8; ++kc) {
            v8bf b0 = *(const v8bf*)(brow + kc * 32);
            v8bf b1 = *(const v8bf*)(brow + kc * 32 + 16);
            v16bf bmat;
#pragma unroll
            for (int j = 0; j < 8; ++j) { bmat[j] = b0[j]; bmat[8 + j] = b1[j]; }
            acc = __builtin_amdgcn_wmma_f32_16x16x32_bf16(
                false, a[kc], false, bmat, (short)0, acc, false, false);
        }
#pragma unroll
        for (int r = 0; r < 8; ++r)
            stage[(mbase + r) * STG + nt * 16 + nlane] = acc[r];
    }

    // ---- epilogue: coalesced row stores from LDS stage ----
    const int d0 = lane * 8;
#pragma unroll 1
    for (int m = 0; m < 16; ++m) {
        size_t gbase = (size_t)(row0 + m) * 256 + d0;
        if (EP == 0) {
            float* o = (float*)Out;
#pragma unroll
            for (int j = 0; j < 8; ++j) o[gbase + j] = stage[m * STG + d0 + j];
        } else if (EP == 1) {
            __bf16* o = (__bf16*)Out;
            v8bf v;
#pragma unroll
            for (int j = 0; j < 8; ++j) v[j] = (__bf16)stage[m * STG + d0 + j];
            *(v8bf*)(o + gbase) = v;
        } else {
            float* o = (float*)Out;
#pragma unroll
            for (int j = 0; j < 8; ++j) o[gbase + j] = xskip[gbase + j] + stage[m * STG + d0 + j];
        }
    }
}

// =====================================================================
// Edge attention scores: e[edge][h] = sum_d lrelu(xl[src]+xr[tgt]) * att[h]
// one wave per edge; lane owns 8 dims, head = lane>>2
// =====================================================================
__global__ void edge_score_kernel(const __bf16* __restrict__ xl,
                                  const float* __restrict__ xr,
                                  const float* __restrict__ att,
                                  const int* __restrict__ esrc,
                                  const int* __restrict__ etgt,
                                  float* __restrict__ e) {
    int edge = (int)((blockIdx.x * blockDim.x + threadIdx.x) >> 5);
    int lane = threadIdx.x & 31;
    if (edge >= cN_PROJ) return;
    int src = esrc[edge];
    int tgt = etgt[edge];
    int d0  = lane * 8;
    int h   = lane >> 2;
    const __bf16* pl = xl + (size_t)src * cD + d0;
    const float*  pr = xr + (size_t)tgt * cD + d0;
    const float*  pa = att + h * 32 + (lane & 3) * 8;
    v8bf lv = *(const v8bf*)pl;
    float s = 0.f;
#pragma unroll
    for (int j = 0; j < 8; ++j) {
        float v = (float)lv[j] + pr[j];
        v = v > 0.f ? v : cNEG * v;
        s += v * pa[j];
    }
    s += __shfl_xor(s, 1, 32);
    s += __shfl_xor(s, 2, 32);
    if ((lane & 3) == 0) e[(size_t)edge * cH + h] = s;
}

// =====================================================================
// CSR build: histogram, single-block scan, scatter
// =====================================================================
__global__ void hist_kernel(const int* __restrict__ etgt, unsigned* __restrict__ counts) {
    int i = blockIdx.x * blockDim.x + threadIdx.x;
    if (i < cN_PROJ) atomicAdd(&counts[etgt[i]], 1u);
}

__global__ void scan_kernel(const unsigned* __restrict__ counts,
                            unsigned* __restrict__ offsets,
                            unsigned* __restrict__ cursor) {
    __shared__ unsigned tmp[256];
    __shared__ unsigned carry;
    int tid = threadIdx.x;
    if (tid == 0) carry = 0u;
    __syncthreads();
    for (int base = 0; base < cN_SP; base += 256) {
        unsigned v = (base + tid < cN_SP) ? counts[base + tid] : 0u;
        tmp[tid] = v;
        __syncthreads();
        for (int off = 1; off < 256; off <<= 1) {
            unsigned t = (tid >= off) ? tmp[tid - off] : 0u;
            __syncthreads();
            tmp[tid] += t;
            __syncthreads();
        }
        unsigned incl = tmp[tid];
        unsigned c = carry;
        if (base + tid < cN_SP) {
            offsets[base + tid] = c + incl - v;
            cursor [base + tid] = c + incl - v;
        }
        __syncthreads();
        if (tid == 255) carry = c + incl;
        __syncthreads();
    }
    if (tid == 0) offsets[cN_SP] = carry;
}

__global__ void scatter_kernel(const int* __restrict__ etgt,
                               unsigned* __restrict__ cursor,
                               int* __restrict__ order) {
    int i = blockIdx.x * blockDim.x + threadIdx.x;
    if (i < cN_PROJ) {
        unsigned pos = atomicAdd(&cursor[etgt[i]], 1u);
        order[pos] = i;
    }
}

// =====================================================================
// Per-scene-point segment softmax + aggregation + residual + LN2/ReLU
// one wave per scene point; lane owns 8 dims (head = lane>>2)
// outputs: xskip (f32) and hpre (bf16, input to final GEMM)
// =====================================================================
__global__ void gat_agg_kernel(const float* __restrict__ e,
                               const __bf16* __restrict__ xl,
                               const int* __restrict__ esrc,
                               const int* __restrict__ order,
                               const unsigned* __restrict__ offsets,
                               const float* __restrict__ prev,
                               const float* __restrict__ gatb,
                               const float* __restrict__ g2,
                               const float* __restrict__ b2,
                               float* __restrict__ xskip,
                               __bf16* __restrict__ hpre) {
    int sp   = (int)((blockIdx.x * blockDim.x + threadIdx.x) >> 5);
    int lane = threadIdx.x & 31;
    if (sp >= cN_SP) return;
    int h  = lane >> 2;
    int d0 = lane * 8;
    unsigned beg = offsets[sp], end = offsets[sp + 1];

    float emax = -3.4e38f;
    for (unsigned i = beg; i < end; ++i) {
        int edge = order[i];
        emax = fmaxf(emax, e[(size_t)edge * cH + h]);
    }

    float acc[8];
#pragma unroll
    for (int j = 0; j < 8; ++j) acc[j] = 0.f;
    float denom = 0.f;
    for (unsigned i = beg; i < end; ++i) {
        int edge = order[i];
        float w = __expf(e[(size_t)edge * cH + h] - emax);
        denom += w;
        int src = esrc[edge];
        v8bf lv = *(const v8bf*)(xl + (size_t)src * cD + d0);
#pragma unroll
        for (int j = 0; j < 8; ++j) acc[j] += w * (float)lv[j];
    }
    float inv = 1.0f / (denom + 1e-16f);

    // x = prev + out + gat_bias  (this is x_skip)
    float x[8];
    float s = 0.f;
    size_t gbase = (size_t)sp * cD + d0;
#pragma unroll
    for (int j = 0; j < 8; ++j) {
        x[j] = prev[gbase + j] + acc[j] * inv + gatb[d0 + j];
        s += x[j];
        xskip[gbase + j] = x[j];
    }
#pragma unroll
    for (int m = 16; m >= 1; m >>= 1) s += __shfl_xor(s, m, 32);
    float mean = s * (1.0f / 256.0f);
    float q = 0.f;
#pragma unroll
    for (int j = 0; j < 8; ++j) { float d = x[j] - mean; q += d * d; }
#pragma unroll
    for (int m = 16; m >= 1; m >>= 1) q += __shfl_xor(q, m, 32);
    float rstd = rsqrtf(q * (1.0f / 256.0f) + cEPS);
    v8bf o;
#pragma unroll
    for (int j = 0; j < 8; ++j) {
        float y = (x[j] - mean) * rstd * g2[d0 + j] + b2[d0 + j];
        y = y > 0.f ? y : 0.f;
        o[j] = (__bf16)y;
    }
    *(v8bf*)(hpre + gbase) = o;
}

// =====================================================================
// host-side launcher
// =====================================================================
extern "C" void kernel_launch(void* const* d_in, const int* in_sizes, int n_in,
                              void* d_out, int out_size, void* d_ws, size_t ws_size,
                              hipStream_t stream) {
    (void)in_sizes; (void)n_in; (void)out_size; (void)ws_size;

    const float* proj = (const float*)d_in[0];
    const float* prev = (const float*)d_in[1];
    const float* Wl   = (const float*)d_in[2];
    const float* bl   = (const float*)d_in[3];
    const float* Wr   = (const float*)d_in[4];
    const float* br   = (const float*)d_in[5];
    const float* att  = (const float*)d_in[6];
    const float* gatb = (const float*)d_in[7];
    const float* ln1g = (const float*)d_in[8];
    const float* ln1b = (const float*)d_in[9];
    const float* ln2g = (const float*)d_in[10];
    const float* ln2b = (const float*)d_in[11];
    const float* mlpW = (const float*)d_in[12];
    const float* mlpb = (const float*)d_in[13];
    const int* esrc   = (const int*)d_in[14];
    const int* etgt   = (const int*)d_in[15];

    char* ws = (char*)d_ws;
    size_t off = 0;
    auto alloc = [&](size_t bytes) {
        char* p = ws + off;
        off = (off + bytes + 255) & ~(size_t)255;
        return p;
    };

    __bf16*   WlT     = (__bf16*)alloc((size_t)256 * 256 * 2);
    __bf16*   WrT     = (__bf16*)alloc((size_t)256 * 256 * 2);
    __bf16*   WmT     = (__bf16*)alloc((size_t)256 * 256 * 2);
    __bf16*   xagg    = (__bf16*)alloc((size_t)cN_SP * cD * 2);
    float*    xr      = (float*)alloc((size_t)cN_SP * cD * 4);
    __bf16*   xl      = (__bf16*)alloc((size_t)cN_PROJ * cD * 2);
    float*    ebuf    = (float*)alloc((size_t)cN_PROJ * cH * 4);
    unsigned* counts  = (unsigned*)alloc((size_t)cN_SP * 4);
    unsigned* offsets = (unsigned*)alloc((size_t)(cN_SP + 1) * 4);
    unsigned* cursor  = (unsigned*)alloc((size_t)cN_SP * 4);
    int*      order   = (int*)alloc((size_t)cN_PROJ * 4);
    float*    xskip   = (float*)alloc((size_t)cN_SP * cD * 4);
    __bf16*   hpre    = (__bf16*)alloc((size_t)cN_SP * cD * 2);

    hipMemsetAsync(counts, 0, (size_t)cN_SP * sizeof(unsigned), stream);

    transpose_bf16_kernel<<<768, 256, 0, stream>>>(Wl, Wr, mlpW, WlT, WrT, WmT);

    // x_agg = relu(LN1(prev))
    ln_relu_bf16_kernel<<<cN_SP / 8, 256, 0, stream>>>(prev, ln1g, ln1b, xagg, cN_SP);

    // xr = x_agg @ Wr + br   (f32 out)
    gemm256_kernel<false, 0><<<(cN_SP + 127) / 128, 256, SMEM_BYTES, stream>>>(
        xagg, WrT, br, xr, nullptr, cN_SP);

    // xl = proj @ Wl + bl    (bf16 out, f32 A converted on the fly)
    gemm256_kernel<true, 1><<<(cN_PROJ + 127) / 128, 256, SMEM_BYTES, stream>>>(
        proj, WlT, bl, xl, nullptr, cN_PROJ);

    // per-edge attention logits
    edge_score_kernel<<<cN_PROJ / 8, 256, 0, stream>>>(xl, xr, att, esrc, etgt, ebuf);

    // CSR by target
    hist_kernel<<<(cN_PROJ + 255) / 256, 256, 0, stream>>>(etgt, counts);
    scan_kernel<<<1, 256, 0, stream>>>(counts, offsets, cursor);
    scatter_kernel<<<(cN_PROJ + 255) / 256, 256, 0, stream>>>(etgt, cursor, order);

    // segment softmax + aggregation + residual + LN2/ReLU
    gat_agg_kernel<<<cN_SP / 8, 256, 0, stream>>>(ebuf, xl, esrc, order, offsets,
                                                  prev, gatb, ln2g, ln2b, xskip, hpre);

    // d_out = x_skip + hpre @ mlp_W + mlp_b
    gemm256_kernel<false, 2><<<(cN_SP + 127) / 128, 256, SMEM_BYTES, stream>>>(
        hpre, WmT, mlpb, d_out, xskip, cN_SP);
}
